// GCNLayer_1400159338837
// MI455X (gfx1250) — matrix operand order repaired and checked by reading, not compiled, not run on hardware
//
#include <hip/hip_runtime.h>

// ---------------- CDNA5 WMMA types ----------------
typedef __attribute__((ext_vector_type(16))) __bf16 v16bf;
typedef __attribute__((ext_vector_type(8)))  float  v8f;

__device__ __forceinline__ __bf16 f32_to_bf16(float f) {
    union { float f; unsigned u; } v; v.f = f;
    unsigned r = v.u + 0x7FFFu + ((v.u >> 16) & 1u);   // round-to-nearest-even
    unsigned short h = (unsigned short)(r >> 16);
    __bf16 b;
    __builtin_memcpy(&b, &h, sizeof(h));
    return b;
}

__device__ __forceinline__ void pack8(v16bf& v, int base, float4 a, float4 b) {
    v[base + 0] = f32_to_bf16(a.x); v[base + 1] = f32_to_bf16(a.y);
    v[base + 2] = f32_to_bf16(a.z); v[base + 3] = f32_to_bf16(a.w);
    v[base + 4] = f32_to_bf16(b.x); v[base + 5] = f32_to_bf16(b.y);
    v[base + 6] = f32_to_bf16(b.z); v[base + 7] = f32_to_bf16(b.w);
}

// Build one 16x32 bf16 A-fragment for this lane from an LDS row of f32.
// Per ISA layout the lane's 16 K-values are two contiguous 8-float runs.
__device__ __forceinline__ v16bf load_a_frag(const float* arow, int k0, int half) {
    const float4* p1 = (const float4*)(arow + k0 + half * 8);
    const float4* p2 = (const float4*)(arow + k0 + 16 + half * 8);
    float4 x0 = p1[0], x1 = p1[1];
    float4 y0 = p2[0], y1 = p2[1];
    v16bf a;
    pack8(a, 0, x0, x1);
    pack8(a, 8, y0, y1);
    return a;
}

// ---------------- generic float4 copy (agg1 init = self-loop term) -----------
__global__ void copy_f4_kernel(const float4* __restrict__ src,
                               float4* __restrict__ dst, long n4) {
    long i = (long)blockIdx.x * blockDim.x + threadIdx.x;
    if (i < n4) dst[i] = src[i];
}

// ---------------- edge scatter: agg[dst] += h[src], 4 feats / thread ---------
template <int DF>
__global__ void scatter_edges_kernel(const float* __restrict__ h,
                                     const int* __restrict__ edges,
                                     float* __restrict__ agg, int E) {
    constexpr int G = DF / 4;
    long tid = (long)blockIdx.x * blockDim.x + threadIdx.x;
    if (tid >= (long)E * G) return;
    int e = (int)(tid / G);
    int g = (int)(tid % G);
    int src = edges[2 * e];
    int dst = edges[2 * e + 1];
    const float4* hv = (const float4*)(h + (long)src * DF) + g;
    float4 v = *hv;
    float* p = agg + (long)dst * DF + g * 4;
    atomicAdd(p + 0, v.x);
    atomicAdd(p + 1, v.y);
    atomicAdd(p + 2, v.z);
    atomicAdd(p + 3, v.w);
}

// ---------------- W pre-swizzle into fragment-major bf16 ---------------------
// Wpre[kt][ct][lane][e]: lane's 16 B-fragment elements contiguous (32 bytes).
// B 32x16 bf16 layout: lane = half*16 + n, element e -> K = kt*32 + (e < 8 ? half*8 + e : 16 + half*8 + (e - 8)).
// Columns >= ncols are zero-padded (no col guard in GEMM hot loop).
__global__ void preswizzle_w_kernel(const float* __restrict__ W, __bf16* __restrict__ Wpre,
                                    int ldw, int ncols, int ktiles, int ctiles) {
    int t = blockIdx.x * blockDim.x + threadIdx.x;
    int total = ktiles * ctiles * 32;
    if (t >= total) return;
    int lane = t & 31;
    int ct   = (t >> 5) % ctiles;
    int kt   = (t >> 5) / ctiles;
    int r = lane & 15, half = lane >> 4;
    int col = ct * 16 + r;
    bool colOK = col < ncols;
    int colc = colOK ? col : 0;
    __bf16* outp = Wpre + (size_t)t * 16;
    #pragma unroll
    for (int e = 0; e < 16; ++e) {
        int k = kt * 32 + (e < 8 ? half * 8 + e : 16 + half * 8 + (e - 8));
        float wv = W[(size_t)k * ldw + colc];
        outp[e] = f32_to_bf16(colOK ? wv : 0.0f);
    }
}

// ---------------- GEMM1: X = relu(agg1[Nx64] @ W1[64x128] + b1) --------------
// 256 threads = 8 waves; block covers 16 rows x 128 cols, wave = one 16x16 tile.
// A tile (16x64 f32, contiguous 4KB) staged in LDS once per block.
// Also writes X into agg2 (self-loop init for layer 2).
__global__ void gemm1_relu_kernel(const float* __restrict__ A,        // N x 64
                                  const __bf16* __restrict__ Wpre,    // [2][8][32][16]
                                  const float* __restrict__ bias,     // 128
                                  float* __restrict__ X,              // N x 128
                                  float* __restrict__ agg2,           // N x 128
                                  int nrows) {
    __shared__ __align__(16) float As[16 * 64];
    const int t    = threadIdx.x;
    const int lane = t & 31;
    const int wave = t >> 5;                  // 0..7 -> column tile
    const int r = lane & 15, half = lane >> 4;
    const int row0 = blockIdx.x * 16;

    // cooperative LDS fill: 256 float4 = 16 rows x 64 f32 (contiguous in A)
    {
        long f4    = (long)row0 * 16 + t;
        long maxf4 = (long)nrows * 16 - 1;
        if (f4 > maxf4) f4 = maxf4;           // clamp (data unused for pad rows)
        ((float4*)As)[t] = ((const float4*)A)[f4];
    }
    __syncthreads();

    const int col = wave * 16 + r;
    const float* arow = &As[r * 64];

    v8f acc = {};
    #pragma unroll
    for (int kt = 0; kt < 2; ++kt) {
        v16bf a = load_a_frag(arow, kt * 32, half);
        v16bf b = *(const v16bf*)(Wpre + ((size_t)(kt * 8 + wave) * 32 + lane) * 16);
        acc = __builtin_amdgcn_wmma_f32_16x16x32_bf16(
            false, a, false, b, (short)0, acc, false, false);
    }

    const float bv = bias[col];
    float res[8];
    #pragma unroll
    for (int v = 0; v < 8; ++v) {
        float x = acc[v] + bv;
        res[v] = x > 0.0f ? x : 0.0f;
    }

    // C/D layout: VGPR v -> row (v + 8*half); single uniform bounds branch.
    size_t base = (size_t)(row0 + 8 * half) * 128 + col;
    if (row0 + 16 <= nrows) {
        #pragma unroll
        for (int v = 0; v < 8; ++v) {
            X[base + (size_t)v * 128]    = res[v];
            agg2[base + (size_t)v * 128] = res[v];
        }
    } else {
        #pragma unroll
        for (int v = 0; v < 8; ++v) {
            if (row0 + v + 8 * half < nrows) {
                X[base + (size_t)v * 128]    = res[v];
                agg2[base + (size_t)v * 128] = res[v];
            }
        }
    }
}

// ---------------- GEMM2: logits = agg2[Nx128] @ W2[128x40] + b2 --------------
// 96 threads = 3 waves; block covers 16 rows x 48 cols (40 valid, B zero-padded).
// A tile (16x128 f32, contiguous 8KB) staged in LDS.
__global__ void gemm2_bias_kernel(const float* __restrict__ A,        // N x 128
                                  const __bf16* __restrict__ Wpre,    // [4][3][32][16]
                                  const float* __restrict__ bias,     // 40
                                  float* __restrict__ OUT,            // N x 40
                                  int nrows) {
    __shared__ __align__(16) float As[16 * 128];
    const int t    = threadIdx.x;
    const int lane = t & 31;
    const int wave = t >> 5;                  // 0..2
    const int r = lane & 15, half = lane >> 4;
    const int row0 = blockIdx.x * 16;

    // cooperative LDS fill: 512 float4, 96 threads
    {
        long maxf4 = (long)nrows * 32 - 1;
        for (int i = t; i < 512; i += 96) {
            long f4 = (long)row0 * 32 + i;
            if (f4 > maxf4) f4 = maxf4;
            ((float4*)As)[i] = ((const float4*)A)[f4];
        }
    }
    __syncthreads();

    const int col = wave * 16 + r;
    const bool colOK = col < 40;
    const float* arow = &As[r * 128];

    v8f acc = {};
    #pragma unroll
    for (int kt = 0; kt < 4; ++kt) {
        v16bf a = load_a_frag(arow, kt * 32, half);
        v16bf b = *(const v16bf*)(Wpre + ((size_t)(kt * 3 + wave) * 32 + lane) * 16);
        acc = __builtin_amdgcn_wmma_f32_16x16x32_bf16(
            false, a, false, b, (short)0, acc, false, false);
    }

    const float bv = bias[colOK ? col : 0];
    size_t base = (size_t)(row0 + 8 * half) * 40 + col;
    if ((row0 + 16 <= nrows) & colOK) {
        #pragma unroll
        for (int v = 0; v < 8; ++v) {
            OUT[base + (size_t)v * 40] = acc[v] + bv;
        }
    } else {
        #pragma unroll
        for (int v = 0; v < 8; ++v) {
            if (colOK && (row0 + v + 8 * half < nrows)) {
                OUT[base + (size_t)v * 40] = acc[v] + bv;
            }
        }
    }
}

// ---------------- softmax over C=40, one thread per row, in place -----------
__global__ void softmax40_kernel(float* __restrict__ out, int nrows) {
    int row = blockIdx.x * blockDim.x + threadIdx.x;
    if (row >= nrows) return;
    float* p = out + (long)row * 40;
    float vals[40];
    float m = -3.402823466e+38f;
    #pragma unroll
    for (int i = 0; i < 40; ++i) { vals[i] = p[i]; m = fmaxf(m, vals[i]); }
    float s = 0.0f;
    #pragma unroll
    for (int i = 0; i < 40; ++i) { vals[i] = __expf(vals[i] - m); s += vals[i]; }
    float inv = 1.0f / s;
    #pragma unroll
    for (int i = 0; i < 40; ++i) p[i] = vals[i] * inv;
}

// ---------------- host-side orchestration -----------------------------------
extern "C" void kernel_launch(void* const* d_in, const int* in_sizes, int n_in,
                              void* d_out, int out_size, void* d_ws, size_t ws_size,
                              hipStream_t stream) {
    const float* h   = (const float*)d_in[0];  // N x 64
    const int*   adj = (const int*)  d_in[1];  // E x 2
    const float* W1  = (const float*)d_in[2];  // 64 x 128
    const float* b1  = (const float*)d_in[3];  // 128
    const float* W2  = (const float*)d_in[4];  // 128 x 40
    const float* b2  = (const float*)d_in[5];  // 40

    const int N = in_sizes[0] / 64;
    const int E = in_sizes[1] / 2;

    // ws layout (floats): agg1 N*64 | X N*128 | agg2 N*128 | Wpre1 | Wpre2
    float*  agg1  = (float*)d_ws;
    float*  X     = agg1 + (size_t)N * 64;
    float*  agg2  = X    + (size_t)N * 128;
    __bf16* Wpre1 = (__bf16*)(agg2 + (size_t)N * 128);          // 2*8*32*16 bf16
    __bf16* Wpre2 = Wpre1 + 2 * 8 * 32 * 16;                    // 4*3*32*16 bf16
    float*  out   = (float*)d_out;                              // N x 40

    const int BT = 256;

    // ---- pre-swizzle weights to fragment-major bf16 (tiny, L2-resident) ----
    preswizzle_w_kernel<<<(2 * 8 * 32 + BT - 1) / BT, BT, 0, stream>>>(
        W1, Wpre1, 128, 128, 2, 8);
    preswizzle_w_kernel<<<(4 * 3 * 32 + BT - 1) / BT, BT, 0, stream>>>(
        W2, Wpre2, 40, 40, 4, 3);

    // ---- layer 1: init with self-loop term, then edge scatter ----
    {
        long n4 = (long)N * 64 / 4;
        copy_f4_kernel<<<(unsigned)((n4 + BT - 1) / BT), BT, 0, stream>>>(
            (const float4*)h, (float4*)agg1, n4);
        long work = (long)E * (64 / 4);
        scatter_edges_kernel<64><<<(unsigned)((work + BT - 1) / BT), BT, 0, stream>>>(
            h, adj, agg1, E);
    }
    // ---- dense: X = relu(agg1 @ W1 + b1); also seeds agg2 with self-loops ----
    {
        int blocks = (N + 15) / 16;
        gemm1_relu_kernel<<<blocks, 256, 0, stream>>>(agg1, Wpre1, b1, X, agg2, N);
    }
    // ---- layer 2 scatter over X ----
    {
        long work = (long)E * (128 / 4);
        scatter_edges_kernel<128><<<(unsigned)((work + BT - 1) / BT), BT, 0, stream>>>(
            X, adj, agg2, E);
    }
    // ---- dense: logits = agg2 @ W2 + b2, then softmax ----
    {
        int blocks = (N + 15) / 16;
        gemm2_bias_kernel<<<blocks, 96, 0, stream>>>(agg2, Wpre2, b2, out, N);
        softmax40_kernel<<<(N + BT - 1) / BT, BT, 0, stream>>>(out, N);
    }
}
